// GptOssExperts_55542517071891
// MI455X (gfx1250) — matrix operand order repaired and checked by reading, not compiled, not run on hardware
//
#include <hip/hip_runtime.h>

// GPT-OSS MoE experts (E=8, T=4096, H=I=2048), dense-routed.
// Pre-pass: convert X/W1/W2 fp32 -> bf16 in ws (memory-bound, ~27us, negligible
//           vs ~17x compute-bound GEMMs).
// Stage 1:  per-expert GEMM1 (V_WMMA_F32_16X16X32_BF16) + clamped-SiLU/GLU
//           -> bf16 act in ws.
// Stage 2:  per-expert GEMM2 + bias + routing-weighted combine over experts.
// ALL GEMM tile staging uses GLOBAL_LOAD_ASYNC_TO_LDS_B128 (ASYNCcnt):
// no VGPR round-trip, loads for step k+1 issued before the WMMA block of step k,
// single s_wait_asynccnt 0 after compute -> latency fully hidden.

typedef __bf16 bf16_t;
typedef __attribute__((ext_vector_type(16))) __bf16 v16bf;
typedef __attribute__((ext_vector_type(8)))  __bf16 v8bf;
typedef __attribute__((ext_vector_type(4)))  __bf16 v4bf;
typedef __attribute__((ext_vector_type(8)))  float  v8f;

#define NE 8
#define NT 4096
#define NH 2048
#define NI 2048

__device__ __forceinline__ void async_b128(const bf16_t* g, const bf16_t* l) {
    // Generic LDS pointer: addr[31:0] is the LDS byte address (ISA aperture rule).
    const unsigned lo = (unsigned)(unsigned long long)l;
    asm volatile("global_load_async_to_lds_b128 %0, %1, off"
                 :: "v"(lo), "v"(g) : "memory");
}
__device__ __forceinline__ void wait_async0() {
    asm volatile("s_wait_asynccnt 0" ::: "memory");
}

__device__ __forceinline__ v16bf load_a_frag(const bf16_t* Alds, int m, int hl) {
    // A 16x32 bf16 fragment: lane(M=m, half=hl) holds
    // K = {16*hl .. 16*hl+15} (contiguous, matching B)
    const v8bf lo = *(const v8bf*)&Alds[m * 32 + hl * 16];
    const v8bf hi = *(const v8bf*)&Alds[m * 32 + hl * 16 + 8];
    v16bf a;
#pragma unroll
    for (int j = 0; j < 8; ++j) { a[j] = lo[j]; a[8 + j] = hi[j]; }
    return a;
}

__device__ __forceinline__ float act_fn(float gate, float up) {
    gate = fminf(gate, 7.0f);
    up   = fminf(fmaxf(up, -7.0f), 7.0f);
    const float glu = gate / (1.0f + __expf(-1.702f * gate));
    return (up + 1.0f) * glu;
}

// ---------------------------------------------------------------------------
// Pre-pass: fp32 -> bf16 (vectorized, grid-stride)
// ---------------------------------------------------------------------------
__global__ __launch_bounds__(256) void cvt_bf16_kernel(const float* __restrict__ src,
                                                       bf16_t* __restrict__ dst,
                                                       long n4) {
    const long stride = (long)gridDim.x * 256;
    for (long i = (long)blockIdx.x * 256 + threadIdx.x; i < n4; i += stride) {
        const float4 v = *(const float4*)(src + i * 4);
        v4bf b;
        b[0] = (bf16_t)v.x; b[1] = (bf16_t)v.y; b[2] = (bf16_t)v.z; b[3] = (bf16_t)v.w;
        *(v4bf*)(dst + i * 4) = b;
    }
}

// ---------------------------------------------------------------------------
// Kernel 1: gu = X @ W1^T + b1, act = f(gate, up)
// grid (NI/32, NT/128, NE); 8 waves = 4(M:32 rows) x 2(N:16 i-cols); 4 wmma/step
// ---------------------------------------------------------------------------
__global__ __launch_bounds__(256) void gemm1_act_kernel(
    const bf16_t* __restrict__ Xb,   // (T,H) bf16
    const bf16_t* __restrict__ W1b,  // (E,2I,H) bf16
    const float* __restrict__ B1,    // (E,2I) fp32
    bf16_t* __restrict__ act)        // (E,T,I) bf16
{
    __shared__ bf16_t Alds[2][128 * 32];   // row-major (M,K)
    __shared__ bf16_t Bgl[2][32 * 32];     // N-major gate cols
    __shared__ bf16_t Bul[2][32 * 32];     // N-major up cols

    const int e  = blockIdx.z;
    const int t0 = blockIdx.y * 128;
    const int i0 = blockIdx.x * 32;
    const int tid  = threadIdx.x;
    const int wave = tid >> 5, lane = tid & 31;
    const int wm = wave >> 1, wn = wave & 1;
    const int mrow = lane & 15, hl = lane >> 4;

    // staging geometry: A 128x32 (2x b128/thr), B 64 f-rows x 32 (1x b128/thr)
    const int arow = tid >> 2, ac8 = (tid & 3) * 8;
    const int brow = tid >> 2, bc8 = (tid & 3) * 8;
    const int bn = brow & 31;
    const int bfr = 2 * (i0 + bn) + (brow >> 5);      // even f = gate, odd = up
    const bf16_t* gA0 = Xb + (size_t)(t0 + arow) * NH + ac8;
    const bf16_t* gA1 = Xb + (size_t)(t0 + arow + 64) * NH + ac8;
    const bf16_t* gB  = W1b + ((size_t)e * 2 * NI + bfr) * NH + bc8;

    auto issue = [&](int buf, int k0) {
        async_b128(gA0 + k0, &Alds[buf][arow * 32 + ac8]);
        async_b128(gA1 + k0, &Alds[buf][(arow + 64) * 32 + ac8]);
        const bf16_t* lb = (brow < 32) ? &Bgl[buf][bn * 32 + bc8]
                                       : &Bul[buf][bn * 32 + bc8];
        async_b128(gB + k0, lb);
    };

    v8f accg0 = {}, accg1 = {}, accu0 = {}, accu1 = {};

    issue(0, 0);
    wait_async0();
    __syncthreads();

    constexpr int NK = NH / 32;
    for (int kk = 0; kk < NK; ++kk) {
        const int cur = kk & 1;
        if (kk + 1 < NK) issue(cur ^ 1, (kk + 1) * 32);

        const bf16_t* Ab = Alds[cur];
        const v16bf a0 = load_a_frag(Ab + (wm * 32) * 32, mrow, hl);
        const v16bf a1 = load_a_frag(Ab + (wm * 32 + 16) * 32, mrow, hl);
        const v16bf bg = *(const v16bf*)&Bgl[cur][(wn * 16 + mrow) * 32 + hl * 16];
        const v16bf bu = *(const v16bf*)&Bul[cur][(wn * 16 + mrow) * 32 + hl * 16];
        accg0 = __builtin_amdgcn_wmma_f32_16x16x32_bf16(false, a0, false, bg, (short)0, accg0, false, false);
        accg1 = __builtin_amdgcn_wmma_f32_16x16x32_bf16(false, a1, false, bg, (short)0, accg1, false, false);
        accu0 = __builtin_amdgcn_wmma_f32_16x16x32_bf16(false, a0, false, bu, (short)0, accu0, false, false);
        accu1 = __builtin_amdgcn_wmma_f32_16x16x32_bf16(false, a1, false, bu, (short)0, accu1, false, false);

        wait_async0();
        __syncthreads();
    }

    const int i = i0 + wn * 16 + mrow;
    const float gb = B1[(size_t)e * 2 * NI + 2 * i];
    const float ub = B1[(size_t)e * 2 * NI + 2 * i + 1];
#pragma unroll
    for (int r = 0; r < 8; ++r) {
        const int ta = t0 + wm * 32 + r + 8 * hl;     // C layout: vgpr r -> M=r+8*half
        const int tb = ta + 16;
        act[((size_t)e * NT + ta) * NI + i] = (bf16_t)act_fn(accg0[r] + gb, accu0[r] + ub);
        act[((size_t)e * NT + tb) * NI + i] = (bf16_t)act_fn(accg1[r] + gb, accu1[r] + ub);
    }
}

// ---------------------------------------------------------------------------
// Kernel 2: out[t,h] = sum_e rw[t,e] * (act_e @ W2_e^T + b2_e)[t,h]
// grid (NH/64, NT/128); 8 waves = 4(M:32 rows) x 2(N:32 h-cols); 2x2 reg tile,
// 4 wmma/step; expert loop fused, no atomics.
// ---------------------------------------------------------------------------
__global__ __launch_bounds__(256) void gemm2_combine_kernel(
    const bf16_t* __restrict__ act, // (E,T,I) bf16
    const bf16_t* __restrict__ W2b, // (E,H,I) bf16
    const float* __restrict__ B2,   // (E,H)
    const float* __restrict__ RW,   // (T,E)
    float* __restrict__ out)        // (T,H)
{
    __shared__ bf16_t Alds[2][128 * 32];
    __shared__ bf16_t Blds[2][64 * 32];    // N-major (N=h, K=i)

    const int t0 = blockIdx.y * 128;
    const int h0 = blockIdx.x * 64;
    const int tid  = threadIdx.x;
    const int wave = tid >> 5, lane = tid & 31;
    const int wm = wave >> 1, wn = wave & 1;
    const int mrow = lane & 15, hl = lane >> 4;

    const int arow = tid >> 2, ac8 = (tid & 3) * 8;
    const int brow = tid >> 2, bc8 = (tid & 3) * 8;
    const bf16_t* gA0 = act + (size_t)(t0 + arow) * NI + ac8;        // + e*NT*NI + k0
    const bf16_t* gA1 = act + (size_t)(t0 + arow + 64) * NI + ac8;
    const bf16_t* gB  = W2b + (size_t)(h0 + brow) * NI + bc8;        // + e*NH*NI + k0

    auto issue = [&](int buf, int e, int k0) {
        const size_t ea = (size_t)e * NT * NI + k0;
        const size_t eb = (size_t)e * NH * NI + k0;
        async_b128(gA0 + ea, &Alds[buf][arow * 32 + ac8]);
        async_b128(gA1 + ea, &Alds[buf][(arow + 64) * 32 + ac8]);
        async_b128(gB + eb, &Blds[buf][brow * 32 + bc8]);
    };

    v8f accE00 = {}, accE01 = {}, accE10 = {}, accE11 = {};
    v8f accF00 = {}, accF01 = {}, accF10 = {}, accF11 = {};

    constexpr int NK  = NI / 32;     // 64
    constexpr int TOT = NE * NK;     // 512 pipelined (e,k) steps

    issue(0, 0, 0);
    wait_async0();
    __syncthreads();

    const int hA = h0 + wn * 32 + mrow;
    const int hB = hA + 16;

    for (int step = 0; step < TOT; ++step) {
        const int cur = step & 1;
        const int e   = step >> 6;
        if (step + 1 < TOT) {
            issue(cur ^ 1, (step + 1) >> 6, ((step + 1) & (NK - 1)) * 32);
        }

        const bf16_t* Ab = Alds[cur];
        const v16bf a0 = load_a_frag(Ab + (wm * 32) * 32, mrow, hl);
        const v16bf a1 = load_a_frag(Ab + (wm * 32 + 16) * 32, mrow, hl);
        const v16bf b0 = *(const v16bf*)&Blds[cur][(wn * 32 + mrow) * 32 + hl * 16];
        const v16bf b1 = *(const v16bf*)&Blds[cur][(wn * 32 + 16 + mrow) * 32 + hl * 16];
        accE00 = __builtin_amdgcn_wmma_f32_16x16x32_bf16(false, a0, false, b0, (short)0, accE00, false, false);
        accE01 = __builtin_amdgcn_wmma_f32_16x16x32_bf16(false, a0, false, b1, (short)0, accE01, false, false);
        accE10 = __builtin_amdgcn_wmma_f32_16x16x32_bf16(false, a1, false, b0, (short)0, accE10, false, false);
        accE11 = __builtin_amdgcn_wmma_f32_16x16x32_bf16(false, a1, false, b1, (short)0, accE11, false, false);

        if ((step & (NK - 1)) == (NK - 1)) {
            // end of expert e: fold into final accumulators
            const float db0 = B2[(size_t)e * NH + hA];
            const float db1 = B2[(size_t)e * NH + hB];
#pragma unroll
            for (int r = 0; r < 8; ++r) {
                const int ta = t0 + wm * 32 + r + 8 * hl;
                const int tb = ta + 16;
                const float wa = RW[(size_t)ta * NE + e];
                const float wb = RW[(size_t)tb * NE + e];
                accF00[r] += wa * (accE00[r] + db0);
                accF01[r] += wa * (accE01[r] + db1);
                accF10[r] += wb * (accE10[r] + db0);
                accF11[r] += wb * (accE11[r] + db1);
                accE00[r] = 0.0f; accE01[r] = 0.0f;
                accE10[r] = 0.0f; accE11[r] = 0.0f;
            }
        }

        wait_async0();
        __syncthreads();
    }

#pragma unroll
    for (int r = 0; r < 8; ++r) {
        const int ta = t0 + wm * 32 + r + 8 * hl;
        const int tb = ta + 16;
        out[(size_t)ta * NH + hA] = accF00[r];
        out[(size_t)ta * NH + hB] = accF01[r];
        out[(size_t)tb * NH + hA] = accF10[r];
        out[(size_t)tb * NH + hB] = accF11[r];
    }
}

extern "C" void kernel_launch(void* const* d_in, const int* in_sizes, int n_in,
                              void* d_out, int out_size, void* d_ws, size_t ws_size,
                              hipStream_t stream) {
    (void)in_sizes; (void)n_in; (void)out_size; (void)ws_size;
    const float* X  = (const float*)d_in[0];   // hidden_states (T,H)
    const float* RW = (const float*)d_in[1];   // routing_weights (T,E)
    const float* W1 = (const float*)d_in[2];   // gate_up_proj (E,2I,H)
    const float* B1 = (const float*)d_in[3];   // gate_up_bias (E,2I)
    const float* W2 = (const float*)d_in[4];   // down_proj (E,H,I)
    const float* B2 = (const float*)d_in[5];   // down_bias (E,H)
    float* out = (float*)d_out;

    // ws layout (bf16 elements): act | W1b | W2b | Xb   (~336 MiB total)
    bf16_t* base = (bf16_t*)d_ws;
    bf16_t* act = base;                                   // E*T*I
    bf16_t* W1b = act + (size_t)NE * NT * NI;             // E*2I*H
    bf16_t* W2b = W1b + (size_t)NE * 2 * NI * NH;         // E*H*I
    bf16_t* Xb  = W2b + (size_t)NE * NH * NI;             // T*H

    cvt_bf16_kernel<<<8192, 256, 0, stream>>>(X,  Xb,  (long)NT * NH / 4);
    cvt_bf16_kernel<<<8192, 256, 0, stream>>>(W1, W1b, (long)NE * 2 * NI * NH / 4);
    cvt_bf16_kernel<<<8192, 256, 0, stream>>>(W2, W2b, (long)NE * NH * NI / 4);

    dim3 g1(NI / 32, NT / 128, NE);
    gemm1_act_kernel<<<g1, dim3(256), 0, stream>>>(Xb, W1b, B1, act);

    dim3 g2(NH / 64, NT / 128, 1);
    gemm2_combine_kernel<<<g2, dim3(256), 0, stream>>>(act, W2b, B2, RW, out);
}
